// EfficientMultiHeadEncoder_17386027614757
// MI455X (gfx1250) — matrix operand order repaired and checked by reading, not compiled, not run on hardware
//
#include <hip/hip_runtime.h>
#include <hip/hip_bf16.h>
#include <stdint.h>
#include <stddef.h>

// Problem constants (match reference)
#define NN      20000
#define EE      640000
#define IND     128
#define HIDD    256
#define MAXWW   512
#define NHEADS  32
#define BN_EPS  1e-5f

typedef __attribute__((ext_vector_type(16))) __bf16 v16bf;
typedef __attribute__((ext_vector_type(8)))  float  v8f;

union Frag {
  v16bf  v;
  __bf16 e[16];
  uint4  q[2];
};

__device__ __forceinline__ v8f wmma_bf16(v16bf a, v16bf b, v8f c) {
  return __builtin_amdgcn_wmma_f32_16x16x32_bf16(false, a, false, b, (short)0, c,
                                                 false, false);
}

// A fragment (16x32 bf16, row-major source, leading dim ld).
// Lane ph=lane>>4 selects K groups {ph*8..ph*8+7} and {16+ph*8..16+ph*8+7}:
// both are 16B-contiguous -> two b128 loads.
__device__ __forceinline__ v16bf load_a_bf16(const __bf16* src, int ld,
                                             int row0, int k0, int lane) {
  Frag A;
  int r = lane & 15, ph = lane >> 4;
  const __bf16* p = src + (size_t)(row0 + r) * ld + k0 + ph * 8;
  A.q[0] = *(const uint4*)(p);
  A.q[1] = *(const uint4*)(p + 16);
  return A.v;
}

// A fragment from f32 source, converted to bf16 with native cvt.
__device__ __forceinline__ v16bf load_a_f32(const float* src, int ld,
                                            int row0, int k0, int lane) {
  Frag A;
  int r = lane & 15, ph = lane >> 4;
  const float* p = src + (size_t)(row0 + r) * ld + k0 + ph * 8;
  float4 f0 = *(const float4*)(p);
  float4 f1 = *(const float4*)(p + 4);
  float4 f2 = *(const float4*)(p + 16);
  float4 f3 = *(const float4*)(p + 20);
  A.e[0]  = (__bf16)f0.x; A.e[1]  = (__bf16)f0.y;
  A.e[2]  = (__bf16)f0.z; A.e[3]  = (__bf16)f0.w;
  A.e[4]  = (__bf16)f1.x; A.e[5]  = (__bf16)f1.y;
  A.e[6]  = (__bf16)f1.z; A.e[7]  = (__bf16)f1.w;
  A.e[8]  = (__bf16)f2.x; A.e[9]  = (__bf16)f2.y;
  A.e[10] = (__bf16)f2.z; A.e[11] = (__bf16)f2.w;
  A.e[12] = (__bf16)f3.x; A.e[13] = (__bf16)f3.y;
  A.e[14] = (__bf16)f3.z; A.e[15] = (__bf16)f3.w;
  return A.v;
}

// B fragment (32x16 bf16) from TRANSPOSED weights: srcT is [Ncols][K], ld = K.
// Lane holds col N=lane&15, K groups {ph*8..ph*8+7} and {16+ph*8..16+ph*8+7} -> two b128.
__device__ __forceinline__ v16bf load_b_bf16t(const __bf16* srcT, int ld,
                                              int k0, int col0, int lane) {
  Frag B;
  int c = lane & 15, ph = lane >> 4;
  const __bf16* p = srcT + (size_t)(col0 + c) * ld + k0 + ph * 8;
  B.q[0] = *(const uint4*)(p);
  B.q[1] = *(const uint4*)(p + 16);
  return B.v;
}

// ---------------- prep kernels ----------------

// Wb [128][256] f32 -> wbT [256][128] bf16
__global__ void k_prep_wb(const float* Wb, __bf16* wbT) {
  int i = blockIdx.x * blockDim.x + threadIdx.x;          // over 256*128
  if (i >= HIDD * IND) return;
  int n = i / IND, k = i % IND;
  wbT[i] = (__bf16)Wb[(size_t)k * HIDD + n];
}

// W1 [H][256][512] f32 -> w1T [H][512][256] bf16
__global__ void k_prep_w1(const float* W1, __bf16* w1T) {
  size_t i = (size_t)blockIdx.x * blockDim.x + threadIdx.x;   // over H*512*256
  if (i >= (size_t)NHEADS * MAXWW * HIDD) return;
  size_t h = i / ((size_t)MAXWW * HIDD);
  size_t rem = i % ((size_t)MAXWW * HIDD);
  size_t m = rem / HIDD, d = rem % HIDD;
  w1T[i] = (__bf16)W1[(h * HIDD + d) * MAXWW + m];
}

// W2 [H][512][256] f32 -> w2T [H][256][512] bf16
__global__ void k_prep_w2(const float* W2, __bf16* w2T) {
  size_t i = (size_t)blockIdx.x * blockDim.x + threadIdx.x;   // over H*256*512
  if (i >= (size_t)NHEADS * HIDD * MAXWW) return;
  size_t h = i / ((size_t)HIDD * MAXWW);
  size_t rem = i % ((size_t)HIDD * MAXWW);
  size_t d = rem / MAXWW, m = rem % MAXWW;
  w2T[i] = (__bf16)W2[(h * MAXWW + m) * HIDD + d];
}

__global__ void k_zero_stats(float* sum, float* sumsq) {
  int i = blockIdx.x * blockDim.x + threadIdx.x;
  if (i < NHEADS * MAXWW) { sum[i] = 0.f; sumsq[i] = 0.f; }
}

__global__ void k_init_agg(const float* x, float* agg) {
  size_t i = (size_t)blockIdx.x * blockDim.x + threadIdx.x;
  if (i < (size_t)NN * IND) agg[i] = x[i];
}

// per edge: agg[dst] += x[src]; 32 lanes per edge, 4 floats / lane
__global__ void k_edge_scatter(const float* x, const int* ei, float* agg) {
  size_t t = (size_t)blockIdx.x * blockDim.x + threadIdx.x;
  size_t e = t >> 5;
  if (e >= EE) return;
  int f = (int)(t & 31) * 4;
  int src = ei[e];
  int dst = ei[EE + e];
  const float4 v = *(const float4*)(x + (size_t)src * IND + f);
  float* d = agg + (size_t)dst * IND + f;
  unsafeAtomicAdd(d + 0, v.x);
  unsafeAtomicAdd(d + 1, v.y);
  unsafeAtomicAdd(d + 2, v.z);
  unsafeAtomicAdd(d + 3, v.w);
}

// ---------------- backbone: h = relu(agg @ Wb + bb), bf16 out ----------------
// grid (N/16), block 128 (4 waves); wave -> 16x64 tile, block covers all 256 cols
__global__ void k_backbone(const float* agg, const __bf16* wbT,
                           const float* bb, __bf16* h) {
  int wave = threadIdx.x >> 5, lane = threadIdx.x & 31;
  int row0 = blockIdx.x * 16;
  int col0 = wave * 64;
  v8f c[4] = {};
#pragma unroll
  for (int k0 = 0; k0 < IND; k0 += 32) {
    v16bf a = load_a_f32(agg, IND, row0, k0, lane);
#pragma unroll
    for (int t = 0; t < 4; ++t) {
      v16bf b = load_b_bf16t(wbT, IND, k0, col0 + t * 16, lane);
      c[t] = wmma_bf16(a, b, c[t]);
    }
  }
  int colL = lane & 15, ph = lane >> 4;
#pragma unroll
  for (int t = 0; t < 4; ++t) {
    int colg = col0 + t * 16 + colL;
    float bias = bb[colg];
#pragma unroll
    for (int v = 0; v < 8; ++v) {
      float val = fmaxf(c[t][v] + bias, 0.f);
      int row = row0 + v + ph * 8;
      h[(size_t)row * HIDD + colg] = (__bf16)val;
    }
  }
}

// ---------------- GEMM1: z = h @ W1[h] + b1, store bf16 + BN stats ----------------
// grid (N/16, 512/256, 32 heads), block 128; wave -> 16x64 tile
__global__ void k_gemm1(const __bf16* h, const __bf16* w1T,
                        const float* b1, __bf16* z,
                        float* sum, float* sumsq) {
  int wave = threadIdx.x >> 5, lane = threadIdx.x & 31;
  int head = blockIdx.z;
  int row0 = blockIdx.x * 16;
  int col0 = blockIdx.y * 256 + wave * 64;
  const __bf16* bT = w1T + (size_t)head * MAXWW * HIDD;  // [512][256]
  v8f c[4] = {};
#pragma unroll 2
  for (int k0 = 0; k0 < HIDD; k0 += 32) {
    __builtin_prefetch(h + (size_t)(row0 + (lane & 15)) * HIDD + k0 + 32, 0, 0);
    v16bf a = load_a_bf16(h, HIDD, row0, k0, lane);
#pragma unroll
    for (int t = 0; t < 4; ++t) {
      v16bf b = load_b_bf16t(bT, HIDD, k0, col0 + t * 16, lane);
      c[t] = wmma_bf16(a, b, c[t]);
    }
  }
  int colL = lane & 15, ph = lane >> 4;
  __bf16* zh = z + (size_t)head * NN * MAXWW;
  const float* b1h = b1 + head * MAXWW;
#pragma unroll
  for (int t = 0; t < 4; ++t) {
    int colg = col0 + t * 16 + colL;
    float bias = b1h[colg];
    float s = 0.f, q = 0.f;
#pragma unroll
    for (int v = 0; v < 8; ++v) {
      float val = c[t][v] + bias;
      s += val;
      q += val * val;
      int row = row0 + v + ph * 8;
      zh[(size_t)row * MAXWW + colg] = (__bf16)val;
    }
    // combine the two half-waves holding the same column; one atomic per column
    s += __shfl_xor(s, 16, 32);
    q += __shfl_xor(q, 16, 32);
    if (ph == 0) {
      unsafeAtomicAdd(&sum[head * MAXWW + colg], s);
      unsafeAtomicAdd(&sumsq[head * MAXWW + colg], q);
    }
  }
}

// ---------------- BN finalize: scale/shift per (head, m) ----------------
__global__ void k_bn_finalize(const float* sum, const float* sumsq,
                              const float* gamma, const float* beta,
                              float* scale, float* shift) {
  int i = blockIdx.x * blockDim.x + threadIdx.x;
  if (i >= NHEADS * MAXWW) return;
  const float invN = 1.0f / (float)NN;
  float mean = sum[i] * invN;
  float var  = fmaxf(sumsq[i] * invN - mean * mean, 0.f);
  float sc = gamma[i] * rsqrtf(var + BN_EPS);
  scale[i] = sc;
  shift[i] = beta[i] - mean * sc;
}

// ---------------- GEMM2: out[n,h,:] = relu(affine(z)) @ W2[h] + b2 ----------------
// grid (N/16, 1, 32 heads), block 128; wave -> 16x64 tile, block covers all 256 cols
__global__ void k_gemm2(const __bf16* z, const float* scale,
                        const float* shift, const __bf16* w2T,
                        const float* b2, float* out) {
  __shared__ float s_sc[MAXWW];
  __shared__ float s_sh[MAXWW];
  int wave = threadIdx.x >> 5, lane = threadIdx.x & 31;
  int head = blockIdx.z;
  int row0 = blockIdx.x * 16;
  int col0 = wave * 64;
  const __bf16* zh = z + (size_t)head * NN * MAXWW;
  const __bf16* bT = w2T + (size_t)head * HIDD * MAXWW;  // [256][512]

  // stage per-head BN affine params in LDS
  for (int i = threadIdx.x; i < MAXWW; i += blockDim.x) {
    s_sc[i] = scale[head * MAXWW + i];
    s_sh[i] = shift[head * MAXWW + i];
  }
  __syncthreads();

  int r = lane & 15, ph = lane >> 4;
  v8f c[4] = {};
  for (int k0 = 0; k0 < MAXWW; k0 += 32) {
    // A fragment: load bf16 z, normalize + ReLU, convert back to bf16
    Frag Z, A;
    const __bf16* p = zh + (size_t)(row0 + r) * MAXWW + k0 + ph * 8;
    Z.q[0] = *(const uint4*)(p);
    Z.q[1] = *(const uint4*)(p + 16);
    int kbase = k0 + ph * 8;
#pragma unroll
    for (int i = 0; i < 16; ++i) {
      int kg = kbase + (i < 8 ? i : i + 8);
      float f = (float)Z.e[i];
      f = fmaxf(fmaf(f, s_sc[kg], s_sh[kg]), 0.f);
      A.e[i] = (__bf16)f;
    }
#pragma unroll
    for (int t = 0; t < 4; ++t) {
      v16bf b = load_b_bf16t(bT, MAXWW, k0, col0 + t * 16, lane);
      c[t] = wmma_bf16(A.v, b, c[t]);
    }
  }
  int colL = lane & 15;
  const float* b2h = b2 + head * HIDD;
#pragma unroll
  for (int t = 0; t < 4; ++t) {
    int colg = col0 + t * 16 + colL;
    float bias = b2h[colg];
#pragma unroll
    for (int v = 0; v < 8; ++v) {
      int row = row0 + v + ph * 8;
      out[((size_t)row * NHEADS + head) * HIDD + colg] = c[t][v] + bias;
    }
  }
}

// ---------------- launch ----------------

static inline size_t align256(size_t x) { return (x + 255) & ~(size_t)255; }

extern "C" void kernel_launch(void* const* d_in, const int* in_sizes, int n_in,
                              void* d_out, int out_size, void* d_ws, size_t ws_size,
                              hipStream_t stream) {
  const float* x     = (const float*)d_in[0];
  const int*   ei    = (const int*)d_in[1];
  const float* Wb    = (const float*)d_in[2];
  const float* bb    = (const float*)d_in[3];
  const float* W1    = (const float*)d_in[4];
  const float* b1    = (const float*)d_in[5];
  const float* gamma = (const float*)d_in[6];
  const float* beta  = (const float*)d_in[7];
  const float* W2    = (const float*)d_in[8];
  const float* b2    = (const float*)d_in[9];
  float* out = (float*)d_out;

  // workspace carve-up
  char* ws = (char*)d_ws;
  size_t off = 0;
  float*  agg = (float*)(ws + off);  off = align256(off + (size_t)NN * IND * 4);
  __bf16* h   = (__bf16*)(ws + off); off = align256(off + (size_t)NN * HIDD * 2);
  __bf16* wbT = (__bf16*)(ws + off); off = align256(off + (size_t)HIDD * IND * 2);
  __bf16* w1T = (__bf16*)(ws + off); off = align256(off + (size_t)NHEADS * MAXWW * HIDD * 2);
  __bf16* w2T = (__bf16*)(ws + off); off = align256(off + (size_t)NHEADS * HIDD * MAXWW * 2);
  float* sum   = (float*)(ws + off); off = align256(off + (size_t)NHEADS * MAXWW * 4);
  float* sumsq = (float*)(ws + off); off = align256(off + (size_t)NHEADS * MAXWW * 4);
  float* scale = (float*)(ws + off); off = align256(off + (size_t)NHEADS * MAXWW * 4);
  float* shift = (float*)(ws + off); off = align256(off + (size_t)NHEADS * MAXWW * 4);
  __bf16* z = (__bf16*)(ws + off);
  off = align256(off + (size_t)NHEADS * NN * MAXWW * 2);
  (void)off; (void)ws_size; (void)in_sizes; (void)n_in; (void)out_size;

  // weight prep + stat zeroing
  k_prep_wb<<<(HIDD * IND + 255) / 256, 256, 0, stream>>>(Wb, wbT);
  {
    size_t n1 = (size_t)NHEADS * MAXWW * HIDD;
    k_prep_w1<<<(unsigned)((n1 + 255) / 256), 256, 0, stream>>>(W1, w1T);
    k_prep_w2<<<(unsigned)((n1 + 255) / 256), 256, 0, stream>>>(W2, w2T);
  }
  k_zero_stats<<<(NHEADS * MAXWW + 255) / 256, 256, 0, stream>>>(sum, sumsq);

  // aggregation
  k_init_agg<<<(unsigned)(((size_t)NN * IND + 255) / 256), 256, 0, stream>>>(x, agg);
  k_edge_scatter<<<(unsigned)(((size_t)EE * 32 + 255) / 256), 256, 0, stream>>>(x, ei, agg);

  // backbone GEMM (WMMA bf16)
  k_backbone<<<dim3(NN / 16, 1), 128, 0, stream>>>(agg, wbT, bb, h);

  // per-head GEMM1 + BN stats (WMMA bf16)
  k_gemm1<<<dim3(NN / 16, MAXWW / 256, NHEADS), 128, 0, stream>>>(h, w1T, b1, z, sum, sumsq);

  // BN finalize
  k_bn_finalize<<<(NHEADS * MAXWW + 255) / 256, 256, 0, stream>>>(sum, sumsq, gamma, beta,
                                                                  scale, shift);

  // per-head GEMM2 fused affine+ReLU (WMMA bf16)
  k_gemm2<<<dim3(NN / 16, 1, NHEADS), 128, 0, stream>>>(z, scale, shift, w2T, b2, out);
}
